// MANO_28501402976438
// MI455X (gfx1250) — compile-verified, matrix operand order checked by reading, not run on hardware
//
#include <hip/hip_runtime.h>
#include <math.h>

typedef __attribute__((ext_vector_type(16))) _Float16 v16h;
typedef __attribute__((ext_vector_type(8)))  float    v8f;

#define BN      4096
#define MESHN   778
#define KTOT    160     // 10 beta + 135 pose-feature + 15 zero pad
#define KSTEPS  5       // 160 / 32
#define NTILES  49      // ceil(778/16)
#define BTILES  256     // 4096/16

__constant__ int c_parents[16] = {0,0,1,2,0,4,5,0,7,8,0,10,11,0,13,14};
__constant__ int c_tips[5]     = {333,444,672,555,745};

// K index held by (lane, element e) of a 16-bit A/B WMMA fragment (16x16x32).
// ISA 7.12.2: lanes 0-15 VGPR0..3 hold K=0..7, VGPR4..7 hold K=16..23;
// lanes 16-31 hold K=8..15 / K=24..31.
__device__ __forceinline__ int kmap16(int lane, int e) {
  return (e < 8 ? e : e + 8) + ((lane & 16) ? 8 : 0);
}

__device__ __forceinline__ void rodrigues9(float rx, float ry, float rz, float* R) {
  float t2 = rx*rx + ry*ry + rz*rz;
  if (t2 < 1e-24f) {                       // Taylor branch (reference "small")
    float a = 1.0f - t2 * (1.0f/6.0f);
    float b = 0.5f - t2 * (1.0f/24.0f);
    R[0] = 1.0f + b*(rx*rx - t2);  R[1] = -a*rz + b*rx*ry;        R[2] =  a*ry + b*rx*rz;
    R[3] =  a*rz + b*rx*ry;        R[4] = 1.0f + b*(ry*ry - t2);  R[5] = -a*rx + b*ry*rz;
    R[6] = -a*ry + b*rx*rz;        R[7] =  a*rx + b*ry*rz;        R[8] = 1.0f + b*(rz*rz - t2);
  } else {
    float th = sqrtf(t2), inv = 1.0f/th;
    float nx = rx*inv, ny = ry*inv, nz = rz*inv;
    float s = sinf(th), c = cosf(th), u = 1.0f - c;
    R[0] = c + u*nx*nx;      R[1] = u*nx*ny - s*nz;  R[2] = u*nx*nz + s*ny;
    R[3] = u*nx*ny + s*nz;   R[4] = c + u*ny*ny;     R[5] = u*ny*nz - s*nx;
    R[6] = u*nx*nz - s*ny;   R[7] = u*ny*nz + s*nx;  R[8] = c + u*nz*nz;
  }
}

// ---------------------------------------------------------------------------
// Once-per-launch model prep: J0 (16x3) and Jdirs (16x3x10) from J_regressor.
__global__ void k_jreg(const float* __restrict__ Jr, const float* __restrict__ mesh_mu,
                       const float* __restrict__ shapedirs, float* __restrict__ jtab) {
  int idx = blockIdx.x * blockDim.x + threadIdx.x;
  if (idx >= 528) return;
  float acc = 0.0f;
  if (idx < 48) {
    int j = idx / 3, c = idx % 3;
    for (int m = 0; m < MESHN; ++m) acc += Jr[j*MESHN + m] * mesh_mu[m*3 + c];
  } else {
    int q = idx - 48, jc = q / 10, k = q % 10;
    int j = jc / 3, c = jc % 3;
    for (int m = 0; m < MESHN; ++m) acc += Jr[j*MESHN + m] * shapedirs[(m*3 + c)*10 + k];
  }
  jtab[idx] = acc;
}

// Pack D = [shapedirs | posedirs] as f16 WMMA B-fragments: [3ch][49 ntile][5 ks][32 lane][16 e]
__global__ void k_packD(const float* __restrict__ shapedirs, const float* __restrict__ posedirs,
                        _Float16* __restrict__ Dpack) {
  int gid = blockIdx.x * blockDim.x + threadIdx.x;
  const int total = 3 * NTILES * KSTEPS * 512;
  if (gid >= total) return;
  int e    = gid & 15;
  int lane = (gid >> 4) & 31;
  int ks   = (gid >> 9) % KSTEPS;
  int nt   = (gid / (512*KSTEPS)) % NTILES;
  int c    =  gid / (512*KSTEPS*NTILES);
  int n = nt*16 + (lane & 15);
  int K = ks*32 + kmap16(lane, e);
  float v = 0.0f;
  if (n < MESHN) {
    if (K < 10)       v = shapedirs[(n*3 + c)*10 + K];
    else if (K < 145) v = posedirs [(n*3 + c)*135 + (K - 10)];
  }
  Dpack[gid] = (_Float16)v;
}

// Pack skinning weights as B-fragments (K=joint, padded to 32): [49 ntile][32 lane][16 e]
__global__ void k_packW(const float* __restrict__ weights, _Float16* __restrict__ Wpack) {
  int gid = blockIdx.x * blockDim.x + threadIdx.x;
  const int total = NTILES * 512;
  if (gid >= total) return;
  int e    = gid & 15;
  int lane = (gid >> 4) & 31;
  int nt   = gid >> 9;
  int n = nt*16 + (lane & 15);
  int j = kmap16(lane, e);
  float v = (j < 16 && n < MESHN) ? weights[n*16 + j] : 0.0f;
  Wpack[gid] = (_Float16)v;
}

// ---------------------------------------------------------------------------
// Per-batch small math: PCA pose blend, Rodrigues, joints, kinematic chain.
__global__ void k_batch(const float* __restrict__ theta, const float* __restrict__ beta,
                        const float* __restrict__ hc, const float* __restrict__ hmean,
                        const float* __restrict__ jtab,
                        float* __restrict__ Xf, float* __restrict__ Aent,
                        float* __restrict__ Gt, float* __restrict__ Rot) {
  int b = blockIdx.x;
  int t = threadIdx.x;
  __shared__ float s_pose[48];
  __shared__ float s_R[16][9];
  __shared__ float s_J[48];
  __shared__ float s_G[16][16];
  __shared__ float s_beta[10];
  const float* th = theta + (size_t)b*48;

  if (t < 45) {
    float acc = hmean[t];
    for (int p = 0; p < 45; ++p) acc += th[3 + p] * hc[p*45 + t];
    s_pose[3 + t] = acc;
  } else if (t < 48) {
    s_pose[t - 45] = (t == 45) ? 3.14159265358979323846f : 0.0f;  // root rot (pi,0,0)
  } else if (t < 58) {
    s_beta[t - 48] = beta[(size_t)b*10 + (t - 48)];
  }
  __syncthreads();

  if (t < 16) {
    rodrigues9(s_pose[3*t], s_pose[3*t+1], s_pose[3*t+2], s_R[t]);
  } else if (t == 16) {
    float Rr[9];
    rodrigues9(th[0], th[1], th[2], Rr);           // global Rots
    for (int e = 0; e < 9; ++e) Rot[(size_t)b*12 + e] = Rr[e];
  }
  __syncthreads();

  if (t < 48) {                                    // J = J0 + Jdirs @ beta
    float acc = jtab[t];
    for (int k = 0; k < 10; ++k) acc += jtab[48 + t*10 + k] * s_beta[k];
    s_J[t] = acc;
  }
  for (int k = t; k < KTOT; k += 64) {             // x = [beta | R-I | pad]
    float v;
    if (k < 10) v = s_beta[k];
    else if (k < 145) {
      int j = 1 + (k - 10)/9, e = (k - 10)%9;
      v = s_R[j][e] - ((e == 0 || e == 4 || e == 8) ? 1.0f : 0.0f);
    } else v = 0.0f;
    Xf[(size_t)b*KTOT + k] = v;
  }
  __syncthreads();

  if (t == 0) {                                    // kinematic chain
    for (int p = 0; p < 3; ++p) {
      s_G[0][p*4+0] = s_R[0][p*3+0]; s_G[0][p*4+1] = s_R[0][p*3+1];
      s_G[0][p*4+2] = s_R[0][p*3+2]; s_G[0][p*4+3] = s_J[p];
    }
    s_G[0][12] = 0.f; s_G[0][13] = 0.f; s_G[0][14] = 0.f; s_G[0][15] = 1.f;
    for (int i = 1; i < 16; ++i) {
      int pa = c_parents[i];
      float T[16];
      for (int p = 0; p < 3; ++p) {
        T[p*4+0] = s_R[i][p*3+0]; T[p*4+1] = s_R[i][p*3+1]; T[p*4+2] = s_R[i][p*3+2];
        T[p*4+3] = s_J[i*3+p] - s_J[pa*3+p];
      }
      T[12] = 0.f; T[13] = 0.f; T[14] = 0.f; T[15] = 1.f;
      for (int p = 0; p < 4; ++p)
        for (int q = 0; q < 4; ++q) {
          float a = 0.f;
          for (int r2 = 0; r2 < 4; ++r2) a += s_G[pa][p*4+r2] * T[r2*4+q];
          s_G[i][p*4+q] = a;
        }
    }
  }
  __syncthreads();

  for (int idx = t; idx < 192; idx += 64) {        // A = G with corrected last col
    int j = idx/12, pe = idx%12, p = pe/4, q = pe%4;
    float v = s_G[j][p*4+q];
    if (q == 3)
      v -= s_G[j][p*4+0]*s_J[j*3+0] + s_G[j][p*4+1]*s_J[j*3+1] + s_G[j][p*4+2]*s_J[j*3+2];
    Aent[(size_t)b*192 + idx] = v;
  }
  if (t < 48) Gt[(size_t)b*48 + t] = s_G[t/3][(t%3)*4 + 3];
}

// Pack x vectors as f16 A-fragments: [256 btile][5 ks][32 lane][16 e]
__global__ void k_packX(const float* __restrict__ Xf, _Float16* __restrict__ Xpack) {
  int gid = blockIdx.x * blockDim.x + threadIdx.x;
  const int total = BTILES * KSTEPS * 512;
  if (gid >= total) return;
  int e    = gid & 15;
  int lane = (gid >> 4) & 31;
  int ks   = (gid >> 9) % KSTEPS;
  int bt   = gid / (512*KSTEPS);
  int batch = bt*16 + (lane & 15);
  int K = ks*32 + kmap16(lane, e);
  Xpack[gid] = (_Float16)Xf[(size_t)batch*KTOT + K];
}

// Pack A matrix entries as f16 A-fragments (K=joint): [256 btile][12 entry][32 lane][16 e]
__global__ void k_packA(const float* __restrict__ Aent, _Float16* __restrict__ Apack) {
  int gid = blockIdx.x * blockDim.x + threadIdx.x;
  const int total = BTILES * 12 * 512;
  if (gid >= total) return;
  int e    = gid & 15;
  int lane = (gid >> 4) & 31;
  int ent  = (gid >> 9) % 12;
  int bt   = gid / (512*12);
  int batch = bt*16 + (lane & 15);
  int j = kmap16(lane, e);
  Apack[gid] = (_Float16)((j < 16) ? Aent[(size_t)batch*192 + j*12 + ent] : 0.0f);
}

// ---------------------------------------------------------------------------
// Hot kernel: one wave per 16(batch) x 16(vertex) tile.
// 15 WMMAs: blend-shape GEMM (x,y,z channels, K=160). 12 WMMAs: skinning blend.
__global__ void __launch_bounds__(32)
k_vertex(const _Float16* __restrict__ Xpack, const _Float16* __restrict__ Dpack,
         const _Float16* __restrict__ Wpack, const _Float16* __restrict__ Apack,
         const float* __restrict__ Rot, const float* __restrict__ mesh_mu,
         float* __restrict__ out0) {
  int lane = threadIdx.x;
  int nt = blockIdx.x, bt = blockIdx.y;
  int nloc = lane & 15;
  int vtx = nt*16 + nloc;

  // v_posed (pre-bias) via 16x16x32 f16 WMMA, f32 accumulate
  v8f accx = {}, accy = {}, accz = {};
  const _Float16* xb = Xpack + (size_t)bt*KSTEPS*512 + lane*16;
  const _Float16* db = Dpack + lane*16;
#pragma unroll
  for (int ks = 0; ks < KSTEPS; ++ks) {
    v16h a  = *(const v16h*)(xb + (size_t)ks*512);
    v16h b0 = *(const v16h*)(db + ((size_t)(0*NTILES + nt)*KSTEPS + ks)*512);
    v16h b1 = *(const v16h*)(db + ((size_t)(1*NTILES + nt)*KSTEPS + ks)*512);
    v16h b2 = *(const v16h*)(db + ((size_t)(2*NTILES + nt)*KSTEPS + ks)*512);
    accx = __builtin_amdgcn_wmma_f32_16x16x32_f16(false, a, false, b0, (short)0, accx, false, false);
    accy = __builtin_amdgcn_wmma_f32_16x16x32_f16(false, a, false, b1, (short)0, accy, false, false);
    accz = __builtin_amdgcn_wmma_f32_16x16x32_f16(false, a, false, b2, (short)0, accz, false, false);
  }

  // skinning blend T[b,vt][e] = sum_j A[b,j][e] * w[vt,j] via WMMA (K=16 padded to 32)
  v16h wfrag = *(const v16h*)(Wpack + (size_t)nt*512 + lane*16);
  v8f T[12];
#pragma unroll
  for (int e = 0; e < 12; ++e) {
    v16h af = *(const v16h*)(Apack + ((size_t)bt*12 + e)*512 + lane*16);
    v8f z = {};
    T[e] = __builtin_amdgcn_wmma_f32_16x16x32_f16(false, af, false, wfrag, (short)0, z, false, false);
  }

  __shared__ float sRot[16][9];
  for (int i = lane; i < 144; i += 32)
    sRot[i/9][i%9] = Rot[((size_t)bt*16 + i/9)*12 + (i%9)];
  __syncthreads();

  float bx = 0.f, by = 0.f, bz = 0.f;
  if (vtx < MESHN) { bx = mesh_mu[vtx*3+0]; by = mesh_mu[vtx*3+1]; bz = mesh_mu[vtx*3+2]; }

#pragma unroll
  for (int r = 0; r < 8; ++r) {
    int m = r + ((lane >> 4) << 3);                 // C-layout: M = r + 8*(lane>=16)
    float vx = accx[r] + bx, vy = accy[r] + by, vz = accz[r] + bz;
    float px = T[0][r]*vx + T[1][r]*vy + T[2][r]*vz  + T[3][r];
    float py = T[4][r]*vx + T[5][r]*vy + T[6][r]*vz  + T[7][r];
    float pz = T[8][r]*vx + T[9][r]*vy + T[10][r]*vz + T[11][r];
    float ox = sRot[m][0]*px + sRot[m][1]*py + sRot[m][2]*pz;
    float oy = sRot[m][3]*px + sRot[m][4]*py + sRot[m][5]*pz;
    float oz = sRot[m][6]*px + sRot[m][7]*py + sRot[m][8]*pz;
    if (vtx < MESHN) {
      size_t base = ((size_t)(bt*16 + m)*MESHN + vtx)*3;
      out0[base+0] = ox; out0[base+1] = oy; out0[base+2] = oz;
    }
  }
}

// ---------------------------------------------------------------------------
// Joints + tips + 2D projection (21 joints per batch).
__global__ void k_joints(const float* __restrict__ cam, const float* __restrict__ mesh_mu,
                         const float* __restrict__ shapedirs, const float* __restrict__ posedirs,
                         const float* __restrict__ weights,
                         const float* __restrict__ Xf, const float* __restrict__ Aent,
                         const float* __restrict__ Gt, const float* __restrict__ Rot,
                         float* __restrict__ out1, float* __restrict__ out2) {
  int b = blockIdx.x;
  int t = threadIdx.x;
  __shared__ float stip[5][3];
  if (t < 5) {                                     // skinned tip vertices (pre-rotation)
    int vtx = c_tips[t];
    const float* x = Xf + (size_t)b*KTOT;
    float vp[3];
    for (int c = 0; c < 3; ++c) {
      float acc = mesh_mu[vtx*3 + c];
      const float* sd = shapedirs + (size_t)(vtx*3 + c)*10;
      for (int k = 0; k < 10; ++k) acc += sd[k]*x[k];
      const float* pd = posedirs + (size_t)(vtx*3 + c)*135;
      for (int p = 0; p < 135; ++p) acc += pd[p]*x[10 + p];
      vp[c] = acc;
    }
    float Tm[12];
    for (int e = 0; e < 12; ++e) Tm[e] = 0.f;
    const float* Ab = Aent + (size_t)b*192;
    for (int j = 0; j < 16; ++j) {
      float w = weights[vtx*16 + j];
      for (int e = 0; e < 12; ++e) Tm[e] += w * Ab[j*12 + e];
    }
    stip[t][0] = Tm[0]*vp[0] + Tm[1]*vp[1] + Tm[2]*vp[2]  + Tm[3];
    stip[t][1] = Tm[4]*vp[0] + Tm[5]*vp[1] + Tm[6]*vp[2]  + Tm[7];
    stip[t][2] = Tm[8]*vp[0] + Tm[9]*vp[1] + Tm[10]*vp[2] + Tm[11];
  }
  __syncthreads();
  if (t < 21) {
    float p3[3];
    if (t > 0 && (t & 3) == 0) {                   // positions 4,8,12,16,20 are tips
      int ti = t/4 - 1;
      p3[0] = stip[ti][0]; p3[1] = stip[ti][1]; p3[2] = stip[ti][2];
    } else {
      int g = t - t/4;                             // skip over interleaved tips
      p3[0] = Gt[(size_t)b*48 + g*3+0];
      p3[1] = Gt[(size_t)b*48 + g*3+1];
      p3[2] = Gt[(size_t)b*48 + g*3+2];
    }
    const float* R = Rot + (size_t)b*12;
    float o0 = R[0]*p3[0] + R[1]*p3[1] + R[2]*p3[2];
    float o1 = R[3]*p3[0] + R[4]*p3[1] + R[5]*p3[2];
    float o2 = R[6]*p3[0] + R[7]*p3[1] + R[8]*p3[2];
    size_t base1 = ((size_t)b*21 + t)*3;
    out1[base1+0] = o0; out1[base1+1] = o1; out1[base1+2] = o2;
    float c0 = cam[(size_t)b*3+0], c1 = cam[(size_t)b*3+1], c2 = cam[(size_t)b*3+2];
    size_t base2 = ((size_t)b*21 + t)*2;
    out2[base2+0] = c1 + c0*o0;
    out2[base2+1] = c2 + c0*o1;
  }
}

// ---------------------------------------------------------------------------
extern "C" void kernel_launch(void* const* d_in, const int* in_sizes, int n_in,
                              void* d_out, int out_size, void* d_ws, size_t ws_size,
                              hipStream_t stream) {
  const float* theta       = (const float*)d_in[0];
  const float* beta        = (const float*)d_in[1];
  const float* cam         = (const float*)d_in[2];
  const float* mesh_mu     = (const float*)d_in[3];
  const float* shapedirs   = (const float*)d_in[4];
  const float* posedirs    = (const float*)d_in[5];
  const float* J_regressor = (const float*)d_in[6];
  const float* weights     = (const float*)d_in[7];
  const float* hc          = (const float*)d_in[8];
  const float* hmean       = (const float*)d_in[9];

  // Workspace layout (floats first, then 32B-aligned f16 fragment buffers).
  float* fw   = (float*)d_ws;
  float* jtab = fw;                              // 528
  float* Xf   = fw + 528;                        // BN*160
  float* Aent = Xf + (size_t)BN*KTOT;            // BN*192
  float* Gt   = Aent + (size_t)BN*192;           // BN*48
  float* Rot  = Gt + (size_t)BN*48;              // BN*12
  _Float16* Xpack = (_Float16*)(Rot + (size_t)BN*12);     // BTILES*KSTEPS*512
  _Float16* Dpack = Xpack + (size_t)BTILES*KSTEPS*512;    // 3*NTILES*KSTEPS*512
  _Float16* Wpack = Dpack + (size_t)3*NTILES*KSTEPS*512;  // NTILES*512
  _Float16* Apack = Wpack + (size_t)NTILES*512;           // BTILES*12*512
  // total ~= 12.0 MB

  k_jreg<<<(528 + 63)/64, 64, 0, stream>>>(J_regressor, mesh_mu, shapedirs, jtab);
  { int total = 3*NTILES*KSTEPS*512;
    k_packD<<<(total + 255)/256, 256, 0, stream>>>(shapedirs, posedirs, Dpack); }
  { int total = NTILES*512;
    k_packW<<<(total + 255)/256, 256, 0, stream>>>(weights, Wpack); }
  k_batch<<<BN, 64, 0, stream>>>(theta, beta, hc, hmean, jtab, Xf, Aent, Gt, Rot);
  { int total = BTILES*KSTEPS*512;
    k_packX<<<(total + 255)/256, 256, 0, stream>>>(Xf, Xpack); }
  { int total = BTILES*12*512;
    k_packA<<<(total + 255)/256, 256, 0, stream>>>(Aent, Apack); }

  float* out0 = (float*)d_out;                       // x3d[:, 21:]  (B,778,3)
  float* out1 = out0 + (size_t)BN*MESHN*3;           // x3d[:, :21]  (B,21,3)
  float* out2 = out1 + (size_t)BN*21*3;              // x2d[:, :21]  (B,21,2)

  k_vertex<<<dim3(NTILES, BTILES), 32, 0, stream>>>(Xpack, Dpack, Wpack, Apack,
                                                    Rot, mesh_mu, out0);
  k_joints<<<BN, 32, 0, stream>>>(cam, mesh_mu, shapedirs, posedirs, weights,
                                  Xf, Aent, Gt, Rot, out1, out2);
}
